// Attention_58574763983246
// MI455X (gfx1250) — matrix operand order repaired and checked by reading, not compiled, not run on hardware
//
#include <hip/hip_runtime.h>

// ---------------------------------------------------------------------------
// B=8, T=4096, D=256 causal attention (mean-subtraction in reference cancels):
//   Y  = X @ W                      (keys, bf16, ws[0 .. 16MB))
//   Xt = transpose(X) as bf16       (values, [B][D][T], ws[16MB .. 32MB))
//   out[b,j,:] = sum_{i<=j} softmax_i(Y[b,i,:] . X[b,j,:]) * X[b,i,:]
// ---------------------------------------------------------------------------

typedef __attribute__((ext_vector_type(16))) __bf16 v16bf;
typedef __attribute__((ext_vector_type(8)))  __bf16 v8bf;
typedef __attribute__((ext_vector_type(8)))  float  v8f;
typedef __attribute__((ext_vector_type(8)))  float  f32x8;
typedef __attribute__((ext_vector_type(4)))  int    v4i;

#define WMMA_BF16(a, b, c) \
  __builtin_amdgcn_wmma_f32_16x16x32_bf16(false, (a), false, (b), (short)0, (c), false, false)

static __device__ __forceinline__ v16bf bfcat(v8bf lo, v8bf hi) {
  return __builtin_shufflevector(lo, hi, 0,1,2,3,4,5,6,7,8,9,10,11,12,13,14,15);
}

// 16-byte async global->LDS copy (gfx1250, ASYNCcnt-tracked, bypasses VGPRs).
// Offset must be an integer constant expression -> template parameter.
template <int OFF>
static __device__ __forceinline__ void async_cp16(const void* g, void* l) {
  __builtin_amdgcn_global_load_async_to_lds_b128((v4i*)g, (v4i*)l, OFF, 0);
}

#define BATCH 8
#define SEQ   4096
#define DIM   256

// ---------------------------------------------------------------------------
// Kernel 0: Xt[b][d][i] = (bf16) x[b][i][d]   (32x32 LDS tiles, coalesced)
// ---------------------------------------------------------------------------
__global__ __launch_bounds__(256) void xprep_kernel(const float* __restrict__ x,
                                                    __bf16* __restrict__ xt) {
  __shared__ __bf16 tile[32][33];
  const int b = blockIdx.z, i0 = blockIdx.x * 32, d0 = blockIdx.y * 32;
  const int t = threadIdx.x, r = t >> 5, c = t & 31;
  #pragma unroll
  for (int rr = r; rr < 32; rr += 8)
    tile[rr][c] = (__bf16)x[((size_t)b * SEQ + i0 + rr) * DIM + d0 + c];
  __syncthreads();
  #pragma unroll
  for (int rr = r; rr < 32; rr += 8)
    xt[((size_t)b * DIM + d0 + rr) * SEQ + i0 + c] = tile[c][rr];
}

// ---------------------------------------------------------------------------
// Kernel 1: Y[b] = X[b] @ W  -> bf16, row-major (K-dim contiguous for WMMA B).
// ---------------------------------------------------------------------------
__global__ __launch_bounds__(256) void ygemm_kernel(const float* __restrict__ x,
                                                    const float* __restrict__ w,
                                                    __bf16* __restrict__ y) {
  __shared__ __align__(64) __bf16 Wt[16 * 256];  // Wt[n][k] = W[k][d0+n]
  const int t  = threadIdx.x;
  const int d0 = blockIdx.x * 16;

  #pragma unroll
  for (int it = 0; it < 16; ++it) {
    int idx = t + it * 256;
    int k = idx >> 4, nn = idx & 15;
    Wt[nn * 256 + k] = (__bf16)w[k * 256 + d0 + nn];
  }
  __syncthreads();

  const int wv = t >> 5, lane = t & 31, n = lane & 15, half = lane >> 4;
  const int off = half * 8;
  const int m = blockIdx.y * 128 + wv * 16 + n;        // A-operand: lane = M
  const float* xrow = x + (((size_t)blockIdx.z * SEQ) + m) * DIM;

  v8f acc;
  #pragma unroll
  for (int i = 0; i < 8; ++i) acc[i] = 0.f;

  #pragma unroll
  for (int c = 0; c < 8; ++c) {                        // K = 256 = 8 x 32
    f32x8 qlo = *(const f32x8*)(xrow + c * 32 + off);
    f32x8 qhi = *(const f32x8*)(xrow + c * 32 + 16 + off);
    v16bf a;
    #pragma unroll
    for (int i = 0; i < 8; ++i) { a[i] = (__bf16)qlo[i]; a[i + 8] = (__bf16)qhi[i]; }
    v16bf bmat = bfcat(*(const v8bf*)(Wt + n * 256 + c * 32 + off), *(const v8bf*)(Wt + n * 256 + c * 32 + 16 + off));
    acc = WMMA_BF16(a, bmat, acc);
  }

  __bf16* yt = y + (((size_t)blockIdx.z * SEQ) + blockIdx.y * 128 + wv * 16) * DIM + d0;
  #pragma unroll
  for (int r = 0; r < 8; ++r) yt[(r + half * 8) * DIM + n] = (__bf16)acc[r];
}

// ---------------------------------------------------------------------------
// Kernel 2: flash attention, async double-buffered K/V staging.
// 256 threads = 8 waves; 128 query rows/block (16/wave); key blocks of 32.
// ---------------------------------------------------------------------------
__global__ __launch_bounds__(256) void attn_kernel(const float* __restrict__ x,
                                                   const __bf16* __restrict__ y,
                                                   const __bf16* __restrict__ xt,
                                                   float* __restrict__ out) {
  __shared__ __align__(64) __bf16 Ks[2][32 * 256];   // keys row-major (k contiguous)
  __shared__ __align__(64) __bf16 Vt[2][256 * 32];   // values transposed [d][key]
  __shared__ __align__(64) __bf16 Pw[8][16 * 32];    // per-wave P (C->A relayout)

  const int b  = blockIdx.x >> 5;
  const int bm = blockIdx.x & 31;
  const int j0 = bm * 128;
  const int t = threadIdx.x, wv = t >> 5, lane = t & 31, n = lane & 15, half = lane >> 4;
  const int off = half * 8;
  const int j0w = j0 + wv * 16;

  // Q (16 rows per wave) into registers in WMMA A layout, fp32 -> bf16.
  const float* xq = x + (((size_t)b * SEQ) + j0w + n) * DIM;
  v16bf qa[8];
  #pragma unroll
  for (int c = 0; c < 8; ++c) {
    f32x8 qlo = *(const f32x8*)(xq + c * 32 + off);
    f32x8 qhi = *(const f32x8*)(xq + c * 32 + 16 + off);
    #pragma unroll
    for (int i = 0; i < 8; ++i) { qa[c][i] = (__bf16)qlo[i]; qa[c][i + 8] = (__bf16)qhi[i]; }
  }

  v8f O[16];
  #pragma unroll
  for (int dc = 0; dc < 16; ++dc)
    #pragma unroll
    for (int i = 0; i < 8; ++i) O[dc][i] = 0.f;
  float rowmax[8], rowsum[8];
  #pragma unroll
  for (int r = 0; r < 8; ++r) { rowmax[r] = -__builtin_inff(); rowsum[r] = 0.f; }

  // ---- async tile staging: 8 x b128 per thread per stage (16KB K + 16KB V) ----
  const char* gkbase = (const char*)(y  + (((size_t)b * SEQ) + 0) * DIM);
  const char* gvbase = (const char*)(xt + (((size_t)b * DIM) + t) * SEQ);
  auto stage = [&](int buf, int i0) {
    const char* gk = gkbase + (size_t)i0 * DIM * 2 + t * 16;  // 32 rows of Y: 16KB
    char* lk = (char*)Ks[buf] + t * 16;
    async_cp16<0>    (gk, lk);
    async_cp16<4096> (gk, lk);
    async_cp16<8192> (gk, lk);
    async_cp16<12288>(gk, lk);
    const char* gv = gvbase + (size_t)i0 * 2;                 // row d=t of Xt: 64B
    char* lv = (char*)(Vt[buf] + t * 32);
    async_cp16<0> (gv, lv);
    async_cp16<16>(gv, lv);
    async_cp16<32>(gv, lv);
    async_cp16<48>(gv, lv);
  };

  const int nkb = bm * 4 + 4;                          // causal: keys up to j0+127
  stage(0, 0);
  for (int kb = 0; kb < nkb; ++kb) {
    const int cur = kb & 1;
    const int i0  = kb * 32;
    __builtin_amdgcn_s_wait_asynccnt(0);               // current buffer landed
    __syncthreads();                                   // visible; other buffer free
    if (kb + 1 < nkb) stage(cur ^ 1, i0 + 32);         // prefetch next block
    if (i0 > j0w + 15) continue;                       // wave-uniform causal skip

    const __bf16* ks = Ks[cur];
    const __bf16* vt = Vt[cur];

    // S (16x32) = Q . K^T as two 16x16 accumulators over 8 k-chunks
    v8f s0, s1;
    #pragma unroll
    for (int i = 0; i < 8; ++i) { s0[i] = 0.f; s1[i] = 0.f; }
    #pragma unroll
    for (int c = 0; c < 8; ++c) {
      v16bf b0 = bfcat(*(const v8bf*)(ks + n * 256        + c * 32 + off), *(const v8bf*)(ks + n * 256        + c * 32 + 16 + off));
      v16bf b1 = bfcat(*(const v8bf*)(ks + (n + 16) * 256 + c * 32 + off), *(const v8bf*)(ks + (n + 16) * 256 + c * 32 + 16 + off));
      s0 = WMMA_BF16(qa[c], b0, s0);
      s1 = WMMA_BF16(qa[c], b1, s1);
    }

    // Online softmax (rows live in 16-lane groups; 4-step butterflies)
    const int iA = i0 + n, iB = i0 + 16 + n;
    #pragma unroll
    for (int r = 0; r < 8; ++r) {
      const int j = j0w + r + half * 8;
      float v0 = (iA <= j) ? s0[r] : -__builtin_inff();
      float v1 = (iB <= j) ? s1[r] : -__builtin_inff();
      float mt = fmaxf(v0, v1);
      #pragma unroll
      for (int ms = 1; ms < 16; ms <<= 1) mt = fmaxf(mt, __shfl_xor(mt, ms, 32));
      float mn = fmaxf(rowmax[r], mt);
      float al = __expf(rowmax[r] - mn);
      rowmax[r] = mn;
      float p0 = __expf(v0 - mn);
      float p1 = __expf(v1 - mn);
      float rs = p0 + p1;
      #pragma unroll
      for (int ms = 1; ms < 16; ms <<= 1) rs += __shfl_xor(rs, ms, 32);
      rowsum[r] = rowsum[r] * al + rs;
      #pragma unroll
      for (int dc = 0; dc < 16; ++dc) O[dc][r] *= al;
      Pw[wv][(r + half * 8) * 32 + n]      = (__bf16)p0;
      Pw[wv][(r + half * 8) * 32 + 16 + n] = (__bf16)p1;
    }

    // Reload P in A layout (compiler orders via dscnt) and O += P.V
    v8bf plo = *(const v8bf*)(&Pw[wv][n * 32 + off]);
    v8bf phi = *(const v8bf*)(&Pw[wv][n * 32 + 16 + off]);
    v16bf pa = bfcat(plo, phi);
    #pragma unroll
    for (int dc = 0; dc < 16; ++dc) {
      v16bf bv = bfcat(*(const v8bf*)(vt + (dc * 16 + n) * 32 + off), *(const v8bf*)(vt + (dc * 16 + n) * 32 + 16 + off));
      O[dc] = WMMA_BF16(pa, bv, O[dc]);
    }
  }

  // Normalize by row sum and store fp32
  float* op = out + (((size_t)b * SEQ) + j0w) * DIM;
  #pragma unroll
  for (int r = 0; r < 8; ++r) {
    const float inv = 1.0f / rowsum[r];
    #pragma unroll
    for (int dc = 0; dc < 16; ++dc)
      op[(r + half * 8) * DIM + dc * 16 + n] = O[dc][r] * inv;
  }
}

// ---------------------------------------------------------------------------
extern "C" void kernel_launch(void* const* d_in, const int* in_sizes, int n_in,
                              void* d_out, int out_size, void* d_ws, size_t ws_size,
                              hipStream_t stream) {
  (void)in_sizes; (void)n_in; (void)out_size; (void)ws_size;
  const float* x = (const float*)d_in[0];      // [B,T,D] fp32
  const float* w = (const float*)d_in[1];      // [D,D]   fp32
  float* out = (float*)d_out;                  // [B,T,D] fp32
  __bf16* y  = (__bf16*)d_ws;                                   // 16 MiB
  __bf16* xt = (__bf16*)((char*)d_ws + (size_t)BATCH*SEQ*DIM*2); // 16 MiB

  xprep_kernel<<<dim3(SEQ / 32, DIM / 32, BATCH), 256, 0, stream>>>(x, xt);
  ygemm_kernel<<<dim3(DIM / 16, SEQ / 128, BATCH), 256, 0, stream>>>(x, w, y);
  attn_kernel<<<BATCH * (SEQ / 128), 256, 0, stream>>>(x, y, xt, out);
}
